// snn_linear_36129264894778
// MI455X (gfx1250) — matrix operand order repaired and checked by reading, not compiled, not run on hardware
//
#include <hip/hip_runtime.h>
#include <hip/hip_bf16.h>

// ---------------------------------------------------------------------------
// SNN LIF layer for MI455X (gfx1250):
//   1) wconv:  W f32 [K][N] -> bf16 transposed [N][K] in workspace, with the
//              k-permutation (0,2,1,3 per 4-group) matching the A-stage packer
//   2) GEMM:   cur[16384, 2048] = spikes(bin) x W, bf16 WMMA, double-buffered
//              LDS with GLOBAL_LOAD_ASYNC_TO_LDS_B128 for the B tiles
//   3) scan:   LIF mempot recurrence over T (in-place spikes + mempot_final)
//   4) trace:  input trace decay
//
// d_out layout (reference return order, all f32):
//   [0, T*B*NOUT)  spikes_out [T][B][NOUT]
//   [.., +B*NIN)   traces_final
//   [.., +B*NOUT)  mempot_final
// ---------------------------------------------------------------------------

#define SNN_B    256
#define SNN_T    64
#define SNN_NIN  2048
#define SNN_NOUT 2048
#define SNN_M    (SNN_B * SNN_T)          // 16384 GEMM rows
#define DECAYF   0.9375f                  // (2^4-1)/2^4

typedef __attribute__((ext_vector_type(16))) __bf16 v16bf;
typedef __attribute__((ext_vector_type(8)))  float  v8f;

union Frag16 {
    v16bf bf;
    float4 f4[2];
    unsigned short us[16];
};

__device__ __forceinline__ unsigned short f32_to_bf16_rne(float f) {
    unsigned int u = __float_as_uint(f);
    unsigned int r = u + 0x7FFFu + ((u >> 16) & 1u);
    return (unsigned short)(r >> 16);
}

// Low 32 bits of a generic pointer into LDS == LDS byte offset (aperture form).
__device__ __forceinline__ unsigned lds_off32(const void* p) {
    return (unsigned)(uintptr_t)p;
}

// Issue two async 16B global->LDS copies (ASYNCcnt += 2). INST_OFFSET applies
// to both the LDS and the global address, so one (vdst, vaddr) pair suffices.
__device__ __forceinline__ void async_copy_32B(unsigned lds_dst, const void* gsrc) {
    unsigned long long ga = (unsigned long long)(uintptr_t)gsrc;
    asm volatile(
        "global_load_async_to_lds_b128 %0, %1, off\n\t"
        "global_load_async_to_lds_b128 %0, %1, off offset:16"
        :: "v"(lds_dst), "v"(ga) : "memory");
}

__device__ __forceinline__ void wait_async0() {
    asm volatile("s_wait_asynccnt 0x0" ::: "memory");
}

// ---------------------------------------------------------------------------
// Kernel 1: weights f32 [K][N] -> bf16 transposed [N][K], k permuted so that
// position 4g+p holds k = 4g+r with p = ((r&1)<<1)|(r>>1)  (r order 0,2,1,3).
// This matches the A-tile packer below; a GEMM is invariant under any k
// permutation applied identically to A and B.
// ---------------------------------------------------------------------------
__global__ void wconv_kernel(const float* __restrict__ w, unsigned short* __restrict__ wt) {
    int idx = blockIdx.x * blockDim.x + threadIdx.x;       // k*N + n
    int k = idx >> 11;
    int n = idx & (SNN_NOUT - 1);
    int r = k & 3;
    int pos = (k & ~3) + (((r & 1) << 1) | (r >> 1));
    wt[(size_t)n * SNN_NIN + pos] = f32_to_bf16_rne(w[idx]);
}

// ---------------------------------------------------------------------------
// Kernel 2: bf16 WMMA GEMM, block tile 128x128, K-step 32, 8 waves (4x2 grid,
// 32x64 per wave = 2x4 accumulators), double-buffered LDS, async B copies.
// ---------------------------------------------------------------------------
#define LDS_STRIDE 40   // halves per row: 32 data + 8 pad (keeps b128 alignment)

__global__ void __launch_bounds__(256)
snn_gemm_kernel(const unsigned char* __restrict__ spk,
                const unsigned short* __restrict__ wt,
                float* __restrict__ cur_out) {
    __shared__ __align__(16) unsigned short As[2][128 * LDS_STRIDE];
    __shared__ __align__(16) unsigned short Bs[2][128 * LDS_STRIDE];

    const int tid    = threadIdx.x;
    const int lane   = tid & 31;
    const int wv     = tid >> 5;
    const int lane16 = lane & 15;
    const int hi     = lane >> 4;
    const int m_off  = (wv >> 1) * 32;     // wave row: 0,32,64,96
    const int n_off  = (wv & 1) * 64;      // wave col: 0 or 64

    const int n0 = blockIdx.x * 128;
    const int m0 = blockIdx.y * 128;

    v8f acc[2][4];
#pragma unroll
    for (int i = 0; i < 2; ++i)
#pragma unroll
        for (int j = 0; j < 4; ++j)
            acc[i][j] = (v8f)(0.0f);

    // Stage-in mappings (one 128x32 tile each):
    //   A: 128 rows x 32 u8  -> 16 B per thread (packed multiply -> bf16)
    //   B: 128 rows x 32 bf16 -> 32 B per thread via 2 async b128 copies
    const int srow  = tid >> 1;
    const int apart = (tid & 1) * 16;      // u8 columns
    const int bpart = (tid & 1) * 16;      // bf16 columns

    const unsigned char*  a_gbase = spk + (size_t)(m0 + srow) * SNN_NIN + apart;
    const unsigned short* b_gbase = wt  + (size_t)(n0 + srow) * SNN_NIN + bpart;
    unsigned short* a_lbase0 = &As[0][srow * LDS_STRIDE + apart];
    unsigned short* a_lbase1 = &As[1][srow * LDS_STRIDE + apart];
    const unsigned  b_loff0  = lds_off32(&Bs[0][srow * LDS_STRIDE + bpart]);
    const unsigned  b_loff1  = lds_off32(&Bs[1][srow * LDS_STRIDE + bpart]);

    // 16 spike bytes (0/1) -> 16 bf16 halves, two at a time:
    //   (w & 0x00010001) * 0x3F80  -> packed bf16 for bytes {0,2}
    //   (w>>8 & 0x00010001)*0x3F80 -> packed bf16 for bytes {1,3}
    // Emits k order (0,2,1,3) per 4-group, matched by wconv's permutation.
    auto stage_A = [&](unsigned short* dst, uint4 av) {
        const unsigned int w4[4] = {av.x, av.y, av.z, av.w};
        unsigned int o[8];
#pragma unroll
        for (int j = 0; j < 4; ++j) {
            o[2 * j]     = (w4[j] & 0x00010001u) * 0x3F80u;
            o[2 * j + 1] = ((w4[j] >> 8) & 0x00010001u) * 0x3F80u;
        }
        *(uint4*)(dst)     = *(const uint4*)&o[0];
        *(uint4*)(dst + 8) = *(const uint4*)&o[4];
    };

    // ---- prologue: fill stage 0 ------------------------------------------
    async_copy_32B(b_loff0, b_gbase);                 // B tile, kk = 0
    stage_A(a_lbase0, *(const uint4*)a_gbase);        // A tile, kk = 0
    wait_async0();
    __syncthreads();

    unsigned buf = 0;
    for (int kk = 0; kk < SNN_NIN; kk += 32) {
        const bool more = (kk + 32) < SNN_NIN;
        uint4 a_next;
        if (more) {
            // Start next stage's DMA + A load before computing this stage.
            async_copy_32B(buf ? b_loff0 : b_loff1, b_gbase + kk + 32);
            a_next = *(const uint4*)(a_gbase + kk + 32);
        }

        // ---- compute current stage: fragments + 8 WMMAs ------------------
        const unsigned short* as = As[buf];
        const unsigned short* bs = Bs[buf];
        Frag16 afr[2], bfr[4];
#pragma unroll
        for (int mt = 0; mt < 2; ++mt) {
            const unsigned short* ap = &as[(m_off + mt * 16 + lane16) * LDS_STRIDE + hi * 8];
            afr[mt].f4[0] = *(const float4*)(ap);        // k = 8*hi + 0..7
            afr[mt].f4[1] = *(const float4*)(ap + 16);   // k = 16 + 8*hi + 0..7
        }
#pragma unroll
        for (int nt = 0; nt < 4; ++nt) {
            const unsigned short* bp = &bs[(n_off + nt * 16 + lane16) * LDS_STRIDE + hi * 8];
            bfr[nt].f4[0] = *(const float4*)(bp);        // k = 8*hi + 0..7
            bfr[nt].f4[1] = *(const float4*)(bp + 16);   // k = 16 + 8*hi + 0..7
        }
#pragma unroll
        for (int mt = 0; mt < 2; ++mt)
#pragma unroll
            for (int nt = 0; nt < 4; ++nt)
                acc[mt][nt] = __builtin_amdgcn_wmma_f32_16x16x32_bf16(
                    false, afr[mt].bf, false, bfr[nt].bf,
                    (short)0, acc[mt][nt], false, false);

        if (more) {
            stage_A(buf ? a_lbase0 : a_lbase1, a_next);  // ds_store next A
            wait_async0();                               // next B tile in LDS
            __syncthreads();                             // stage handoff
            buf ^= 1;
        }
    }

    // ---- store C, scattering m -> (t,b) into [T][B][NOUT] ----------------
#pragma unroll
    for (int mt = 0; mt < 2; ++mt) {
#pragma unroll
        for (int nt = 0; nt < 4; ++nt) {
            const int n = n0 + n_off + nt * 16 + lane16;
#pragma unroll
            for (int r = 0; r < 8; ++r) {
                const int m = m0 + m_off + mt * 16 + r + hi * 8;
                const int t = m & (SNN_T - 1);
                const int b = m >> 6;                    // m / T
                cur_out[((size_t)t * SNN_B + b) * SNN_NOUT + n] = acc[mt][nt][r];
            }
        }
    }
}

// ---------------------------------------------------------------------------
// Kernel 3: LIF scan over T per (b,n): cur -> spikes (in place) + mempot_final.
// ---------------------------------------------------------------------------
__global__ void lif_scan_kernel(float* __restrict__ out) {
    const int idx = blockIdx.x * blockDim.x + threadIdx.x;   // b*NOUT + n
    const int b = idx >> 11;
    const int n = idx & (SNN_NOUT - 1);
    float mempot = 0.0f;
#pragma unroll 4
    for (int t = 0; t < SNN_T; ++t) {
        const size_t o = ((size_t)t * SNN_B + b) * SNN_NOUT + n;
        mempot += out[o];
        const bool sp = (mempot >= 1.0f);
        out[o] = sp ? 1.0f : 0.0f;
        mempot = sp ? (mempot - 1.0f) : (mempot * DECAYF);
    }
    out[(size_t)SNN_T * SNN_B * SNN_NOUT + (size_t)SNN_B * SNN_NIN + idx] = mempot;
}

// ---------------------------------------------------------------------------
// Kernel 4: input traces: trace = trace*DECAYF + s_t, over T, per (b,i).
// ---------------------------------------------------------------------------
__global__ void trace_kernel(const unsigned char* __restrict__ spk,
                             float* __restrict__ out) {
    const int idx = blockIdx.x * blockDim.x + threadIdx.x;   // b*NIN + i
    const int b = idx >> 11;
    const int i = idx & (SNN_NIN - 1);
    float tr = 0.0f;
#pragma unroll 4
    for (int t = 0; t < SNN_T; ++t)
        tr = tr * DECAYF + (float)spk[((size_t)b * SNN_T + t) * SNN_NIN + i];
    out[(size_t)SNN_T * SNN_B * SNN_NOUT + idx] = tr;
}

// ---------------------------------------------------------------------------
extern "C" void kernel_launch(void* const* d_in, const int* in_sizes, int n_in,
                              void* d_out, int out_size, void* d_ws, size_t ws_size,
                              hipStream_t stream) {
    (void)in_sizes; (void)n_in; (void)out_size; (void)ws_size;
    const unsigned char* spikes  = (const unsigned char*)d_in[0];  // [B][T][NIN] bool bytes
    const float*         weights = (const float*)d_in[1];          // [NIN][NOUT] f32
    float*               out     = (float*)d_out;
    unsigned short*      wt_bf16 = (unsigned short*)d_ws;          // [NOUT][NIN] bf16

    wconv_kernel<<<(SNN_NIN * SNN_NOUT) / 256, 256, 0, stream>>>(weights, wt_bf16);

    dim3 grid(SNN_NOUT / 128, SNN_M / 128);   // (16, 128)
    snn_gemm_kernel<<<grid, 256, 0, stream>>>(spikes, wt_bf16, out);

    lif_scan_kernel<<<(SNN_B * SNN_NOUT) / 256, 256, 0, stream>>>(out);

    trace_kernel<<<(SNN_B * SNN_NIN) / 256, 256, 0, stream>>>(spikes, out);
}
